// PNL_5746666242350
// MI455X (gfx1250) — matrix operand order repaired and checked by reading, not compiled, not run on hardware
//
#include <hip/hip_runtime.h>
#include <hip/hip_bf16.h>

typedef __bf16  bf16_t;
typedef __attribute__((ext_vector_type(16))) __bf16 v16bf;
typedef __attribute__((ext_vector_type(8)))  __bf16 v8bf;
typedef __attribute__((ext_vector_type(8)))  float  v8f;

#define WMMA_BF16(a, b, c) \
  __builtin_amdgcn_wmma_f32_16x16x32_bf16(false, (a), false, (b), (short)0, (c), false, false)

// A-fragment (16x32 bf16): lane<16 holds row=lane, K={0..15}; lane>=16 holds
// row=lane-16, K={16..31}. rowPtr points to K-contiguous bf16 row in LDS.
__device__ __forceinline__ v16bf load_a_frag(const bf16_t* rowPtr, int kbase) {
  union { v16bf v; v8bf h[2]; } u;
  u.h[0] = *reinterpret_cast<const v8bf*>(rowPtr + kbase);
  u.h[1] = *reinterpret_cast<const v8bf*>(rowPtr + kbase + 8);
  return u.v;
}

// gfx1250 async copy: 16B global -> LDS, tracked by ASYNCcnt (CDNA5 ISA).
// LDS byte address = low 32 bits of the generic pointer ({aperture_hi, lds_offset}).
__device__ __forceinline__ void async_copy_b128(void* lds_dst, const void* gsrc) {
  asm volatile("global_load_async_to_lds_b128 %0, %1, off"
               :: "v"((unsigned)(unsigned long long)lds_dst), "v"(gsrc)
               : "memory");
}

// ---------------------------------------------------------------------------
// Kernel 1: projection GEMM  Out[b] (bf16 [128,4096]) = Wm[128,256] @ X[b][256,4096] + bias
// block = 128 threads (4 waves), tile = 64(M) x 64(N), K staged in 128-chunks.
// (fp32 -> bf16 conversion during staging, so VGPR-path staging is required here.)
// ---------------------------------------------------------------------------
__global__ __launch_bounds__(128) void k_proj(const float* __restrict__ X,
                                              const float* __restrict__ Wm,
                                              const float* __restrict__ bias,
                                              bf16_t* __restrict__ Out) {
  const int b   = blockIdx.z;
  const int co0 = blockIdx.y * 64;
  const int n0  = blockIdx.x * 64;
  const int tid = threadIdx.x, lane = tid & 31, wave = tid >> 5;

  __shared__ bf16_t sA[64][272];  // W rows [m][k], full K=256 (544B row = 32B aligned)
  __shared__ bf16_t sB[64][144];  // X^T chunk [n][k], K-chunk=128 (288B row)

  const float* Xb = X + (size_t)b * 256 * 4096;

  for (int idx = tid; idx < 64 * 256; idx += 128) {
    int r = idx >> 8, c = idx & 255;
    sA[r][c] = (bf16_t)Wm[(size_t)(co0 + r) * 256 + c];
  }

  const int halfsel = lane >> 4;
  const int row     = wave * 16 + (lane & 15);
  const int kb      = halfsel ? 16 : 0;
  v8f acc[4] = {};

  for (int nk = 0; nk < 256; nk += 128) {
    __syncthreads();
    for (int idx = tid; idx < 128 * 64; idx += 128) {
      int k = idx >> 6, n = idx & 63;
      sB[n][k] = (bf16_t)Xb[(size_t)(nk + k) * 4096 + n0 + n];
    }
    __syncthreads();
#pragma unroll
    for (int kt = 0; kt < 4; ++kt) {
      const int k0 = kt * 32;
      v16bf av = load_a_frag(&sA[row][0], nk + k0 + kb);
#pragma unroll
      for (int nt = 0; nt < 4; ++nt) {
        int col = nt * 16 + (lane & 15);
        v16bf bv = *reinterpret_cast<const v16bf*>(&sB[col][k0 + halfsel * 16]);
        acc[nt] = WMMA_BF16(av, bv, acc[nt]);
      }
    }
  }

  bf16_t* Ob = Out + (size_t)b * 128 * 4096;
#pragma unroll
  for (int r = 0; r < 8; ++r) {
    const int crow = co0 + wave * 16 + r + halfsel * 8;
    const float bia = bias[crow];
#pragma unroll
    for (int nt = 0; nt < 4; ++nt) {
      int col = n0 + nt * 16 + (lane & 15);
      Ob[(size_t)crow * 4096 + col] = (bf16_t)(acc[nt][r] + bia);
    }
  }
}

// ---------------------------------------------------------------------------
// Kernel 2: kv[b][l][m] = (1/2048) * sum_n Phi[b][l][n] * G[b][m][n]
// Phi,G bf16 [256,2048] per batch. Tile 64x64, K streamed in 64-chunks.
// Both operands are K-contiguous in global AND LDS -> stage them with
// GLOBAL_LOAD_ASYNC_TO_LDS_B128, double-buffered so the async engine copies
// chunk i+1 while WMMA consumes chunk i. 8 B128 copies per thread per chunk,
// so s_wait_asynccnt 8 retires exactly the older buffer (in-order completion).
// ---------------------------------------------------------------------------
__global__ __launch_bounds__(128) void k_kv(const bf16_t* __restrict__ Phi,
                                            const bf16_t* __restrict__ G,
                                            bf16_t* __restrict__ KV) {
  const int b   = blockIdx.z;
  const int l0  = blockIdx.y * 64;
  const int m0c = blockIdx.x * 64;
  const int tid = threadIdx.x, lane = tid & 31, wave = tid >> 5;

  __shared__ bf16_t sP[2][64][80];  // Phi tiles [buf][l][k] (160B row, 16B aligned segs)
  __shared__ bf16_t sG[2][64][80];  // G tiles   [buf][m][k]

  const bf16_t* Pb = Phi + (size_t)b * 256 * 2048;
  const bf16_t* Gb = G   + (size_t)b * 256 * 2048;

  const int halfsel = lane >> 4;
  const int row     = wave * 16 + (lane & 15);
  const int kb      = halfsel ? 16 : 0;
  v8f acc[4] = {};

  // Per-thread copy plan: 1024 B128 vectors per chunk / 128 threads = 8 each.
  auto issue_chunk = [&](int nk, int buf) {
#pragma unroll
    for (int t = 0; t < 8; ++t) {
      const int idx  = tid + t * 128;
      const int tile = idx >> 9;        // 0 = Phi, 1 = G
      const int rem  = idx & 511;
      const int r    = rem >> 3;        // row within tile
      const int seg  = rem & 7;         // 8-element (16B) segment
      if (tile == 0)
        async_copy_b128(&sP[buf][r][seg * 8],
                        &Pb[(size_t)(l0 + r) * 2048 + nk + seg * 8]);
      else
        async_copy_b128(&sG[buf][r][seg * 8],
                        &Gb[(size_t)(m0c + r) * 2048 + nk + seg * 8]);
    }
  };

  issue_chunk(0, 0);  // prologue: chunk 0 -> buffer 0
  int cur = 0;
  for (int nk = 0; nk < 2048; nk += 64) {
    if (nk + 64 < 2048) {
      issue_chunk(nk + 64, cur ^ 1);                    // prefetch next chunk
      asm volatile("s_wait_asynccnt 8" ::: "memory");   // current chunk landed
    } else {
      asm volatile("s_wait_asynccnt 0" ::: "memory");
    }
    __syncthreads();  // all waves' async copies visible
#pragma unroll
    for (int kt = 0; kt < 2; ++kt) {
      const int k0 = kt * 32;
      v16bf av = load_a_frag(&sP[cur][row][0], k0 + kb);
#pragma unroll
      for (int nt = 0; nt < 4; ++nt) {
        int col = nt * 16 + (lane & 15);
        v16bf bv = *reinterpret_cast<const v16bf*>(&sG[cur][col][k0 + halfsel * 16]);
        acc[nt] = WMMA_BF16(av, bv, acc[nt]);
      }
    }
    __syncthreads();  // done reading `cur` before it is overwritten next iter
    cur ^= 1;
  }

  const float scale = 1.0f / 2048.0f;
  bf16_t* Kb = KV + (size_t)b * 256 * 256;
#pragma unroll
  for (int r = 0; r < 8; ++r) {
    const int lr = l0 + wave * 16 + r + halfsel * 8;
#pragma unroll
    for (int nt = 0; nt < 4; ++nt) {
      int col = m0c + nt * 16 + (lane & 15);
      Kb[(size_t)lr * 256 + col] = (bf16_t)(acc[nt][r] * scale);
    }
  }
}

// ---------------------------------------------------------------------------
// Kernel 3: Yt[b][m][n] = sum_k kv[b][k][m] * Theta[b][k][n]   (K=256)
// Yt bf16 [256,2048] per batch (== Y [128,4096] reinterpreted).
// ---------------------------------------------------------------------------
__global__ __launch_bounds__(128) void k_y(const bf16_t* __restrict__ KV,
                                           const bf16_t* __restrict__ Theta,
                                           bf16_t* __restrict__ Y) {
  const int b     = blockIdx.z;
  const int mrow0 = blockIdx.y * 64;
  const int n0    = blockIdx.x * 64;
  const int tid = threadIdx.x, lane = tid & 31, wave = tid >> 5;

  __shared__ bf16_t sKt[64][272];  // kv^T tile [m][k], full K=256
  __shared__ bf16_t sTh[64][144];  // Theta^T chunk [n][k], K-chunk=128

  const bf16_t* Kb = KV    + (size_t)b * 256 * 256;
  const bf16_t* Tb = Theta + (size_t)b * 256 * 2048;

  for (int idx = tid; idx < 64 * 256; idx += 128) {
    int k = idx >> 6, c = idx & 63;
    sKt[c][k] = Kb[(size_t)k * 256 + mrow0 + c];
  }

  const int halfsel = lane >> 4;
  const int row     = wave * 16 + (lane & 15);
  const int kb      = halfsel ? 16 : 0;
  v8f acc[4] = {};

  for (int nk = 0; nk < 256; nk += 128) {
    __syncthreads();
    for (int idx = tid; idx < 128 * 64; idx += 128) {
      int k = idx >> 6, c = idx & 63;
      sTh[c][k] = Tb[(size_t)(nk + k) * 2048 + n0 + c];
    }
    __syncthreads();
#pragma unroll
    for (int kt = 0; kt < 4; ++kt) {
      const int k0 = kt * 32;
      v16bf av = load_a_frag(&sKt[row][0], nk + k0 + kb);
#pragma unroll
      for (int nt = 0; nt < 4; ++nt) {
        int col = nt * 16 + (lane & 15);
        v16bf bv = *reinterpret_cast<const v16bf*>(&sTh[col][k0 + halfsel * 16]);
        acc[nt] = WMMA_BF16(av, bv, acc[nt]);
      }
    }
  }

  bf16_t* Yb = Y + (size_t)b * 256 * 2048;
#pragma unroll
  for (int r = 0; r < 8; ++r) {
    const int mr = mrow0 + wave * 16 + r + halfsel * 8;
#pragma unroll
    for (int nt = 0; nt < 4; ++nt) {
      int col = n0 + nt * 16 + (lane & 15);
      Yb[(size_t)mr * 2048 + col] = (bf16_t)acc[nt][r];
    }
  }
}

// ---------------------------------------------------------------------------
// Shared GEMM core for the W-conv passes:
// acc = Ww[c0+..][0:128] @ Y[b][0:128][n0+..]  (64x64 tile, K=128)
// ---------------------------------------------------------------------------
__device__ __forceinline__ void wconv_tile(const float* __restrict__ Ww,
                                           const bf16_t* __restrict__ Yb,
                                           int c0, int n0, int tid, int lane, int wave,
                                           bf16_t (*sW)[144], bf16_t (*sY)[144],
                                           v8f acc[4]) {
  for (int idx = tid; idx < 64 * 128; idx += 128) {  // sW: coalesced over k
    int c = idx >> 7, k = idx & 127;
    sW[c][k] = (bf16_t)Ww[(size_t)(c0 + c) * 128 + k];
  }
  for (int idx = tid; idx < 64 * 128; idx += 128) {  // sY: coalesced over n
    int k = idx >> 6, c = idx & 63;
    sY[c][k] = Yb[(size_t)k * 4096 + n0 + c];
  }
  __syncthreads();
  const int halfsel = lane >> 4;
  const int row     = wave * 16 + (lane & 15);
  const int kb      = halfsel ? 16 : 0;
#pragma unroll
  for (int kt = 0; kt < 4; ++kt) {
    const int k0 = kt * 32;
    v16bf av = load_a_frag(&sW[row][0], k0 + kb);
#pragma unroll
    for (int nt = 0; nt < 4; ++nt) {
      int col = nt * 16 + (lane & 15);
      v16bf bv = *reinterpret_cast<const v16bf*>(&sY[col][k0 + halfsel * 16]);
      acc[nt] = WMMA_BF16(av, bv, acc[nt]);
    }
  }
}

// Kernel 4: BN stats pass — compute Wy tiles, atomically accumulate sum / sumsq.
__global__ __launch_bounds__(128) void k_wstat(const float* __restrict__ Ww,
                                               const float* __restrict__ Wb,
                                               const bf16_t* __restrict__ Y,
                                               float* __restrict__ bn_sum,
                                               float* __restrict__ bn_sq) {
  const int b  = blockIdx.z;
  const int c0 = blockIdx.y * 64;
  const int n0 = blockIdx.x * 64;
  const int tid = threadIdx.x, lane = tid & 31, wave = tid >> 5;
  __shared__ bf16_t sW[64][144];
  __shared__ bf16_t sY[64][144];
  v8f acc[4] = {};
  wconv_tile(Ww, Y + (size_t)b * 128 * 4096, c0, n0, tid, lane, wave, sW, sY, acc);

  const int halfsel = lane >> 4;
  float s[8], q[8];
#pragma unroll
  for (int r = 0; r < 8; ++r) {
    const int cr = c0 + wave * 16 + r + halfsel * 8;
    const float bia = Wb[cr];
    float ss = 0.f, qq = 0.f;
#pragma unroll
    for (int nt = 0; nt < 4; ++nt) { float v = acc[nt][r] + bia; ss += v; qq += v * v; }
    s[r] = ss; q[r] = qq;
  }
#pragma unroll
  for (int off = 1; off < 16; off <<= 1) {  // reduce within each 16-lane half
#pragma unroll
    for (int r = 0; r < 8; ++r) {
      s[r] += __shfl_xor(s[r], off, 32);
      q[r] += __shfl_xor(q[r], off, 32);
    }
  }
  if ((lane & 15) == 0) {
#pragma unroll
    for (int r = 0; r < 8; ++r) {
      const int cr = c0 + wave * 16 + r + halfsel * 8;
      atomicAdd(&bn_sum[cr], s[r]);
      atomicAdd(&bn_sq[cr],  q[r]);
    }
  }
}

// Kernel 5: final pass — recompute Wy tile, BatchNorm, residual, fp32 out.
__global__ __launch_bounds__(128) void k_wfinal(const float* __restrict__ Ww,
                                                const float* __restrict__ Wb,
                                                const bf16_t* __restrict__ Y,
                                                const float* __restrict__ bn_sum,
                                                const float* __restrict__ bn_sq,
                                                const float* __restrict__ gamma,
                                                const float* __restrict__ beta,
                                                const float* __restrict__ xh,
                                                float* __restrict__ out) {
  const int b  = blockIdx.z;
  const int c0 = blockIdx.y * 64;
  const int n0 = blockIdx.x * 64;
  const int tid = threadIdx.x, lane = tid & 31, wave = tid >> 5;
  __shared__ bf16_t sW[64][144];
  __shared__ bf16_t sY[64][144];
  v8f acc[4] = {};
  wconv_tile(Ww, Y + (size_t)b * 128 * 4096, c0, n0, tid, lane, wave, sW, sY, acc);

  const int halfsel = lane >> 4;
  const float inv_cnt = 1.0f / 65536.0f;  // B*H*W
#pragma unroll
  for (int r = 0; r < 8; ++r) {
    const int cr   = c0 + wave * 16 + r + halfsel * 8;
    const float bia  = Wb[cr];
    const float mean = bn_sum[cr] * inv_cnt;
    const float var  = bn_sq[cr] * inv_cnt - mean * mean;
    const float sc   = rsqrtf(var + 1e-5f) * gamma[cr];
    const float bb   = beta[cr];
#pragma unroll
    for (int nt = 0; nt < 4; ++nt) {
      const int col = n0 + nt * 16 + (lane & 15);
      const size_t off = ((size_t)b * 256 + cr) * 4096 + col;
      out[off] = (acc[nt][r] + bia - mean) * sc + bb + xh[off];
    }
  }
}

__global__ void k_zero(float* __restrict__ p, int n) {
  int i = blockIdx.x * blockDim.x + threadIdx.x;
  if (i < n) p[i] = 0.f;
}

// ---------------------------------------------------------------------------
extern "C" void kernel_launch(void* const* d_in, const int* in_sizes, int n_in,
                              void* d_out, int out_size, void* d_ws, size_t ws_size,
                              hipStream_t stream) {
  const float* x_h     = (const float*)d_in[0];
  const float* x_l     = (const float*)d_in[1];
  const float* g_w     = (const float*)d_in[2];
  const float* g_b     = (const float*)d_in[3];
  const float* theta_w = (const float*)d_in[4];
  const float* theta_b = (const float*)d_in[5];
  const float* phi_w   = (const float*)d_in[6];
  const float* phi_b   = (const float*)d_in[7];
  const float* W_w     = (const float*)d_in[8];
  const float* W_b     = (const float*)d_in[9];
  const float* bn_g    = (const float*)d_in[10];
  const float* bn_b    = (const float*)d_in[11];
  float* out = (float*)d_out;

  const size_t SZ_PROJ = (size_t)16 * 128 * 4096;  // elements per projection buffer
  bf16_t* theta = (bf16_t*)d_ws;
  bf16_t* phi   = theta + SZ_PROJ;
  bf16_t* g     = phi + SZ_PROJ;
  bf16_t* kv    = g + SZ_PROJ;
  bf16_t* y     = kv + (size_t)16 * 256 * 256;
  float*  stats = (float*)(y + SZ_PROJ);  // 512 floats: sum[256] | sumsq[256]

  k_zero<<<dim3(2), dim3(256), 0, stream>>>(stats, 512);

  dim3 gProj(64, 2, 16);
  k_proj<<<gProj, 128, 0, stream>>>(x_h, theta_w, theta_b, theta);
  k_proj<<<gProj, 128, 0, stream>>>(x_l, phi_w, phi_b, phi);
  k_proj<<<gProj, 128, 0, stream>>>(x_l, g_w, g_b, g);

  k_kv<<<dim3(4, 4, 16), 128, 0, stream>>>(phi, g, kv);
  k_y<<<dim3(32, 4, 16), 128, 0, stream>>>(kv, theta, y);

  dim3 gW(64, 4, 16);
  k_wstat<<<gW, 128, 0, stream>>>(W_w, W_b, y, stats, stats + 256);
  k_wfinal<<<gW, 128, 0, stream>>>(W_w, W_b, y, stats, stats + 256, bn_g, bn_b, x_h, out);
}
